// EncoderLayer_67894843015471
// MI455X (gfx1250) — compile-verified, matrix operand order checked by reading, not run on hardware
//
#include <hip/hip_runtime.h>
#include <hip/hip_bf16.h>

#define DM   1024
#define NH   16
#define DKH  64
#define DFF  4096
#define BB   4
#define SS   2048
#define MR   (BB*SS)   // 8192 rows

typedef __hip_bfloat16 bf16;
typedef __attribute__((ext_vector_type(16))) __bf16 v16bf;
typedef __attribute__((ext_vector_type(8)))  float  v8f;

union Frag16 { v16bf v; unsigned u[8]; };
union Frag32 { v8f   v; float    f[8]; };

// K-offset of the packed bf16 pair held in frag VGPR v2 for this lane-half.
// ISA 7.12.2 (16-bit A 16x32): V0..3 hold K={0..7 | 8..15}, V4..7 hold K={16..23 | 24..31}
__device__ __forceinline__ int frag_koff(int v2, int hi) {
  return ((v2 & 3) * 2) + ((v2 >> 2) * 16) + hi * 8;
}

// Low 32 bits of a flat shared-aperture address == LDS byte offset
// (ISA 10.2: LDS offset is the low 32 bits; aperture tag lives in addr[63:32]).
__device__ __forceinline__ unsigned lds_addr32(const void* p) {
  return (unsigned)(size_t)p;
}

// Async global -> LDS copy, 16B per lane, tracked by ASYNCcnt.
__device__ __forceinline__ void lds_async_b128(unsigned lds, const void* gptr) {
  asm volatile("global_load_async_to_lds_b128 %0, %1, off"
               :: "v"(lds), "v"(gptr) : "memory");
}
__device__ __forceinline__ void wait_async_le4() {
  asm volatile("s_wait_asynccnt 0x4" ::: "memory");
}
__device__ __forceinline__ void wait_async_0() {
  asm volatile("s_wait_asynccnt 0x0" ::: "memory");
}

// ---------------------------------------------------------------- weight transpose
// in: fp32 [K][N] row-major  ->  out: bf16 [N][K] row-major
__global__ __launch_bounds__(256)
void cvt_transpose_bf16(const float* __restrict__ in, bf16* __restrict__ out,
                        int K, int N) {
  __shared__ float t[32][33];
  int n0 = blockIdx.x * 32, k0 = blockIdx.y * 32;
  int tx = threadIdx.x & 31, ty = threadIdx.x >> 5;  // 32 x 8
#pragma unroll
  for (int i = 0; i < 32; i += 8)
    t[ty + i][tx] = in[(size_t)(k0 + ty + i) * N + n0 + tx];
  __syncthreads();
#pragma unroll
  for (int i = 0; i < 32; i += 8)
    out[(size_t)(n0 + ty + i) * K + k0 + tx] = __float2bfloat16(t[tx][ty + i]);
}

// ---------------------------------------------------------------- layernorm
// Tensor1DNorm: unbiased std (ddof=1), eps added to std. Emits bf16.
__global__ __launch_bounds__(256)
void layernorm_bf16(const float* __restrict__ x, const float* __restrict__ alpha,
                    const float* __restrict__ beta, bf16* __restrict__ out) {
  int row = blockIdx.x;
  const float* xr = x + (size_t)row * DM;
  int tid = threadIdx.x;
  float v4[4];
  float s = 0.f, sq = 0.f;
#pragma unroll
  for (int i = 0; i < 4; ++i) {
    float t = xr[tid + i * 256];
    v4[i] = t; s += t; sq += t * t;
  }
#pragma unroll
  for (int o = 16; o > 0; o >>= 1) {
    s  += __shfl_xor(s,  o, 32);
    sq += __shfl_xor(sq, o, 32);
  }
  __shared__ float rs[8], rq[8];
  int wid = tid >> 5, lane = tid & 31;
  if (lane == 0) { rs[wid] = s; rq[wid] = sq; }
  __syncthreads();
  s = 0.f; sq = 0.f;
#pragma unroll
  for (int w = 0; w < 8; ++w) { s += rs[w]; sq += rq[w]; }
  float mean = s * (1.0f / DM);
  float var  = (sq - (float)DM * mean * mean) * (1.0f / (DM - 1));
  var = var < 0.f ? 0.f : var;
  float inv = 1.0f / (sqrtf(var) + 1e-6f);
#pragma unroll
  for (int i = 0; i < 4; ++i) {
    int c = tid + i * 256;
    out[(size_t)row * DM + c] =
        __float2bfloat16(alpha[c] * (v4[i] - mean) * inv + beta[c]);
  }
}

// ---------------------------------------------------------------- GEMM
#define BM 128
#define BN 128
#define BK 32
#define GEMM_MODE_QKV    0   // bf16 out permuted to [b,h,s,d]
#define GEMM_MODE_QKV_VT 1   // bf16 out permuted to [b,h,d,s] (for V)
#define GEMM_MODE_RELU   2   // bf16 out, relu
#define GEMM_MODE_RES    3   // f32 out, + residual

// 128 rows x 32 bf16 tile, row stride `stride` elems, via 2 async b128/thread.
__device__ __forceinline__ void copy_tile128x32_async(
    const bf16* g, unsigned ldsbase, int stride, int tid) {
#pragma unroll
  for (int i = 0; i < 2; ++i) {
    int c = tid + i * 256;
    int r = c >> 2;
    int co = (c & 3) * 8;
    lds_async_b128(ldsbase + (unsigned)(r * BK + co) * 2,
                   g + (size_t)r * stride + co);
  }
}

// A: bf16 [M][K] row-major activations.  BwT: bf16 [N][K] (pre-transposed weights).
__global__ __launch_bounds__(256)
void gemm_bf16_wmma(const bf16* __restrict__ A, const bf16* __restrict__ BwT,
                    const float* __restrict__ bias, const float* __restrict__ res,
                    void* __restrict__ out, int M, int N, int K, int mode) {
  __shared__ __align__(16) bf16 As[2][BM * BK];
  __shared__ __align__(16) bf16 Bs[2][BN * BK];   // [n][k]
  const int tid  = threadIdx.x;
  const int lane = tid & 31;
  const int wid  = tid >> 5;
  const int wm = wid >> 2;   // 0..1  (64 rows each)
  const int wn = wid & 3;    // 0..3  (32 cols each)
  const int lm = lane & 15;
  const int hi = lane >> 4;
  const int m0g = blockIdx.y * BM;
  const int n0g = blockIdx.x * BN;

  const bf16* Ag = A   + (size_t)m0g * K;
  const bf16* Bg = BwT + (size_t)n0g * K;
  unsigned asb[2] = { lds_addr32(&As[0][0]), lds_addr32(&As[1][0]) };
  unsigned bsb[2] = { lds_addr32(&Bs[0][0]), lds_addr32(&Bs[1][0]) };

  int kb[8];
#pragma unroll
  for (int v2 = 0; v2 < 8; ++v2) kb[v2] = frag_koff(v2, hi);

  Frag32 acc[4][2];
#pragma unroll
  for (int mt = 0; mt < 4; ++mt)
#pragma unroll
    for (int nt = 0; nt < 2; ++nt)
#pragma unroll
      for (int r = 0; r < 8; ++r) acc[mt][nt].f[r] = 0.f;

  const int kTiles = K / BK;
  // prologue: tile 0 -> buf 0
  copy_tile128x32_async(Ag, asb[0], K, tid);
  copy_tile128x32_async(Bg, bsb[0], K, tid);

  for (int t = 0; t < kTiles; ++t) {
    const int cur = t & 1;
    if (t + 1 < kTiles) {   // overlap: issue tile t+1 before retiring tile t
      copy_tile128x32_async(Ag + (t + 1) * BK, asb[cur ^ 1], K, tid);
      copy_tile128x32_async(Bg + (t + 1) * BK, bsb[cur ^ 1], K, tid);
      wait_async_le4();     // in-order: oldest 4 (= tile t) have landed
    } else {
      wait_async_0();
    }
    __syncthreads();        // all waves' copies of buf `cur` visible

    const bf16* AsC = As[cur];
    const bf16* BsC = Bs[cur];
    Frag16 af[4];
#pragma unroll
    for (int mt = 0; mt < 4; ++mt) {
      int mr2 = wm * 64 + mt * 16 + lm;
#pragma unroll
      for (int v2 = 0; v2 < 8; ++v2)
        af[mt].u[v2] = *(const unsigned*)&AsC[mr2 * BK + kb[v2]];
    }
    Frag16 bfv[2];
#pragma unroll
    for (int nt = 0; nt < 2; ++nt) {
      int nc = wn * 32 + nt * 16 + lm;
#pragma unroll
      for (int v2 = 0; v2 < 8; ++v2)
        bfv[nt].u[v2] = *(const unsigned*)&BsC[nc * BK + kb[v2]];
    }
#pragma unroll
    for (int mt = 0; mt < 4; ++mt)
#pragma unroll
      for (int nt = 0; nt < 2; ++nt)
        acc[mt][nt].v = __builtin_amdgcn_wmma_f32_16x16x32_bf16(
            false, af[mt].v, false, bfv[nt].v, (short)0, acc[mt][nt].v,
            false, false);
    __syncthreads();        // LDS reads retired before buf reuse at t+2
  }

  // epilogue
#pragma unroll
  for (int mt = 0; mt < 4; ++mt) {
#pragma unroll
    for (int nt = 0; nt < 2; ++nt) {
      int nc = n0g + wn * 32 + nt * 16 + lm;
      float bval = bias[nc];
#pragma unroll
      for (int r = 0; r < 8; ++r) {
        int mr2 = m0g + wm * 64 + mt * 16 + r + hi * 8;
        float val = acc[mt][nt].f[r] + bval;
        if (mode == GEMM_MODE_QKV) {
          int b = mr2 >> 11;          // row / S
          int s = mr2 & (SS - 1);
          int h = nc >> 6;
          int d = nc & 63;
          ((bf16*)out)[((((size_t)b * NH + h) * SS + s) << 6) + d] =
              __float2bfloat16(val);
        } else if (mode == GEMM_MODE_QKV_VT) {
          int b = mr2 >> 11;
          int s = mr2 & (SS - 1);
          int h = nc >> 6;
          int d = nc & 63;
          ((bf16*)out)[(((size_t)b * NH + h) * DKH + d) * SS + s] =
              __float2bfloat16(val);
        } else if (mode == GEMM_MODE_RELU) {
          ((bf16*)out)[(size_t)mr2 * N + nc] =
              __float2bfloat16(val > 0.f ? val : 0.f);
        } else {
          ((float*)out)[(size_t)mr2 * N + nc] =
              val + res[(size_t)mr2 * N + nc];
        }
      }
    }
  }
}

// ---------------------------------------------------------------- flash attention
#define QB    128   // queries per block (8 waves x 16)
#define KTILE 64    // keys per iteration

// 64 rows x 64 bf16 tile, row stride `stride`, via 2 async b128/thread.
__device__ __forceinline__ void copy_tile64x64_async(
    const bf16* g, unsigned ldsbase, int stride, int tid) {
#pragma unroll
  for (int i = 0; i < 2; ++i) {
    int c = tid + i * 256;
    int r = c >> 3;
    int co = (c & 7) * 8;
    lds_async_b128(ldsbase + (unsigned)(r * KTILE + co) * 2,
                   g + (size_t)r * stride + co);
  }
}

// Q,K: [b,h,s,d].  V: [b,h,d,s] (pre-transposed by its projection GEMM).
__global__ __launch_bounds__(256)
void flash_attn_wmma(const bf16* __restrict__ Q, const bf16* __restrict__ Kg,
                     const bf16* __restrict__ Vg, const int* __restrict__ mask,
                     bf16* __restrict__ O) {
  __shared__ __align__(16) bf16 Ks[2][KTILE * DKH];   // [key][d]
  __shared__ __align__(16) bf16 Vt[2][DKH * KTILE];   // [d][key]
  __shared__ __align__(16) bf16 Pt[8][16 * KTILE];    // per-wave P [qrow][key]
  const int tid  = threadIdx.x;
  const int lane = tid & 31;
  const int wid  = tid >> 5;
  const int lm = lane & 15;
  const int hi = lane >> 4;
  const int bh = blockIdx.y;        // b*NH + h
  const int bb = bh >> 4;
  const int hh = bh & 15;
  const int q0 = blockIdx.x * QB + wid * 16;

  const bf16* Qh = Q  + (size_t)bh * SS * DKH;
  const bf16* Kh = Kg + (size_t)bh * SS * DKH;
  const bf16* Vh = Vg + (size_t)bh * DKH * SS;   // row d, stride SS
  unsigned ksb[2] = { lds_addr32(&Ks[0][0]), lds_addr32(&Ks[1][0]) };
  unsigned vsb[2] = { lds_addr32(&Vt[0][0]), lds_addr32(&Vt[1][0]) };

  int kb[8];
#pragma unroll
  for (int v2 = 0; v2 < 8; ++v2) kb[v2] = frag_koff(v2, hi);

  // Q fragments straight from global (row-major matches A-frag pair packing)
  Frag16 qf[2];
#pragma unroll
  for (int ks2 = 0; ks2 < 2; ++ks2)
#pragma unroll
    for (int v2 = 0; v2 < 8; ++v2)
      qf[ks2].u[v2] =
          *(const unsigned*)&Qh[(size_t)(q0 + lm) * DKH + ks2 * 32 + kb[v2]];

  float m_i[8], l_i[8];
  Frag32 o[4];
#pragma unroll
  for (int r = 0; r < 8; ++r) { m_i[r] = -3.0e38f; l_i[r] = 0.f; }
#pragma unroll
  for (int dt = 0; dt < 4; ++dt)
#pragma unroll
    for (int r = 0; r < 8; ++r) o[dt].f[r] = 0.f;

  const int nIter = SS / KTILE;
  // prologue: tile 0 -> buf 0
  copy_tile64x64_async(Kh, ksb[0], DKH, tid);
  copy_tile64x64_async(Vh, vsb[0], SS, tid);

  for (int t = 0; t < nIter; ++t) {
    const int cur = t & 1;
    const int ks = t * KTILE;
    if (t + 1 < nIter) {
      copy_tile64x64_async(Kh + (size_t)(ks + KTILE) * DKH, ksb[cur ^ 1], DKH, tid);
      copy_tile64x64_async(Vh + (ks + KTILE),               vsb[cur ^ 1], SS,  tid);
      wait_async_le4();
    } else {
      wait_async_0();
    }
    __syncthreads();
    const bf16* KsC = Ks[cur];
    const bf16* VtC = Vt[cur];

    // scores: 16x64 per wave, k-dim 64 = 2 WMMA steps
    Frag32 sc[4];
#pragma unroll
    for (int nt = 0; nt < 4; ++nt) {
#pragma unroll
      for (int r = 0; r < 8; ++r) sc[nt].f[r] = 0.f;
#pragma unroll
      for (int ks2 = 0; ks2 < 2; ++ks2) {
        Frag16 kf;
#pragma unroll
        for (int v2 = 0; v2 < 8; ++v2)
          kf.u[v2] =
              *(const unsigned*)&KsC[(nt * 16 + lm) * DKH + ks2 * 32 + kb[v2]];
        sc[nt].v = __builtin_amdgcn_wmma_f32_16x16x32_bf16(
            false, qf[ks2].v, false, kf.v, (short)0, sc[nt].v, false, false);
      }
      int key = ks + nt * 16 + lm;       // C-layout: lane holds one column(key)
      float bias_m = (mask[bb * SS + key] == 0) ? -1.0e9f : 0.0f;
#pragma unroll
      for (int r = 0; r < 8; ++r)
        sc[nt].f[r] = sc[nt].f[r] * 0.125f + bias_m;   // 1/sqrt(64)
    }

    // online softmax: row stats live in half-wave (16 lanes hold the 16 cols)
    float bm[8];
#pragma unroll
    for (int r = 0; r < 8; ++r) {
      float t2 = sc[0].f[r];
#pragma unroll
      for (int nt = 1; nt < 4; ++nt) t2 = fmaxf(t2, sc[nt].f[r]);
      bm[r] = t2;
    }
#pragma unroll
    for (int off = 1; off < 16; off <<= 1)
#pragma unroll
      for (int r = 0; r < 8; ++r)
        bm[r] = fmaxf(bm[r], __shfl_xor(bm[r], off, 32));
    float nm[8], al[8];
#pragma unroll
    for (int r = 0; r < 8; ++r) {
      nm[r] = fmaxf(m_i[r], bm[r]);
      al[r] = __expf(m_i[r] - nm[r]);
      m_i[r] = nm[r];
    }
    float bs[8];
#pragma unroll
    for (int r = 0; r < 8; ++r) bs[r] = 0.f;
#pragma unroll
    for (int nt = 0; nt < 4; ++nt)
#pragma unroll
      for (int r = 0; r < 8; ++r) {
        float p = __expf(sc[nt].f[r] - nm[r]);
        sc[nt].f[r] = p;
        bs[r] += p;
      }
#pragma unroll
    for (int off = 1; off < 16; off <<= 1)
#pragma unroll
      for (int r = 0; r < 8; ++r) bs[r] += __shfl_xor(bs[r], off, 32);
#pragma unroll
    for (int r = 0; r < 8; ++r) l_i[r] = l_i[r] * al[r] + bs[r];
#pragma unroll
    for (int dt = 0; dt < 4; ++dt)
#pragma unroll
      for (int r = 0; r < 8; ++r) o[dt].f[r] *= al[r];

    // P (C-layout) -> per-wave LDS -> A-layout fragments; same-wave LDS is
    // in-order via DScnt.
#pragma unroll
    for (int nt = 0; nt < 4; ++nt)
#pragma unroll
      for (int r = 0; r < 8; ++r)
        Pt[wid][(r + hi * 8) * KTILE + nt * 16 + lm] =
            __float2bfloat16(sc[nt].f[r]);

#pragma unroll
    for (int ks2 = 0; ks2 < 2; ++ks2) {
      Frag16 pf;
#pragma unroll
      for (int v2 = 0; v2 < 8; ++v2)
        pf.u[v2] = *(const unsigned*)&Pt[wid][lm * KTILE + ks2 * 32 + kb[v2]];
#pragma unroll
      for (int dt = 0; dt < 4; ++dt) {
        Frag16 vf;
#pragma unroll
        for (int v2 = 0; v2 < 8; ++v2)
          vf.u[v2] =
              *(const unsigned*)&VtC[(dt * 16 + lm) * KTILE + ks2 * 32 + kb[v2]];
        o[dt].v = __builtin_amdgcn_wmma_f32_16x16x32_bf16(
            false, pf.v, false, vf.v, (short)0, o[dt].v, false, false);
      }
    }
    __syncthreads();   // LDS reads retired before buf reuse at t+2
  }

  // O /= l, store as bf16 [b, s, h*64+d]
#pragma unroll
  for (int dt = 0; dt < 4; ++dt) {
#pragma unroll
    for (int r = 0; r < 8; ++r) {
      int s = q0 + r + hi * 8;
      float val = o[dt].f[r] / l_i[r];
      O[((size_t)bb * SS + s) * DM + hh * DKH + dt * 16 + lm] =
          __float2bfloat16(val);
    }
  }
}

// ---------------------------------------------------------------- launch
extern "C" void kernel_launch(void* const* d_in, const int* in_sizes, int n_in,
                              void* d_out, int out_size, void* d_ws, size_t ws_size,
                              hipStream_t stream) {
  (void)in_sizes; (void)n_in; (void)out_size; (void)ws_size;
  const float* x    = (const float*)d_in[0];
  const int*   mask = (const int*)d_in[1];
  const float* wq = (const float*)d_in[2];
  const float* bq = (const float*)d_in[3];
  const float* wk = (const float*)d_in[4];
  const float* bk = (const float*)d_in[5];
  const float* wv = (const float*)d_in[6];
  const float* bv = (const float*)d_in[7];
  const float* wo = (const float*)d_in[8];
  const float* bo = (const float*)d_in[9];
  const float* w1 = (const float*)d_in[10];
  const float* b1 = (const float*)d_in[11];
  const float* w2 = (const float*)d_in[12];
  const float* b2 = (const float*)d_in[13];
  const float* a1  = (const float*)d_in[14];
  const float* be1 = (const float*)d_in[15];
  const float* a2  = (const float*)d_in[16];
  const float* be2 = (const float*)d_in[17];

  char* ws = (char*)d_ws;
  const size_t MBy = 1ull << 20;
  bf16* wqT = (bf16*)(ws + 0   * MBy);   // [DM][DM]  2 MB
  bf16* wkT = (bf16*)(ws + 2   * MBy);   // 2 MB
  bf16* wvT = (bf16*)(ws + 4   * MBy);   // 2 MB
  bf16* woT = (bf16*)(ws + 6   * MBy);   // 2 MB
  bf16* w1T = (bf16*)(ws + 8   * MBy);   // [DFF][DM] 8 MB
  bf16* w2T = (bf16*)(ws + 16  * MBy);   // [DM][DFF] 8 MB
  bf16* x2   = (bf16*)(ws + 24  * MBy);  // 16 MB (reused for both norms)
  bf16* qb   = (bf16*)(ws + 40  * MBy);  // 16 MB [b,h,s,d]
  bf16* kbuf = (bf16*)(ws + 56  * MBy);  // 16 MB [b,h,s,d]
  bf16* vbuf = (bf16*)(ws + 72  * MBy);  // 16 MB [b,h,d,s]
  bf16* attn = (bf16*)(ws + 88  * MBy);  // 16 MB [b,s,h*d]
  bf16* ffh  = (bf16*)(ws + 104 * MBy);  // 64 MB
  float* y1  = (float*)(ws + 168 * MBy); // 32 MB (x + attn projection)

  // weight transposes (once per launch): [K][N] f32 -> [N][K] bf16
  cvt_transpose_bf16<<<dim3(DM / 32, DM / 32),  256, 0, stream>>>(wq, wqT, DM, DM);
  cvt_transpose_bf16<<<dim3(DM / 32, DM / 32),  256, 0, stream>>>(wk, wkT, DM, DM);
  cvt_transpose_bf16<<<dim3(DM / 32, DM / 32),  256, 0, stream>>>(wv, wvT, DM, DM);
  cvt_transpose_bf16<<<dim3(DM / 32, DM / 32),  256, 0, stream>>>(wo, woT, DM, DM);
  cvt_transpose_bf16<<<dim3(DFF / 32, DM / 32), 256, 0, stream>>>(w1, w1T, DM, DFF);
  cvt_transpose_bf16<<<dim3(DM / 32, DFF / 32), 256, 0, stream>>>(w2, w2T, DFF, DM);

  // ---- attention sublayer ----
  layernorm_bf16<<<MR, 256, 0, stream>>>(x, a1, be1, x2);
  dim3 g1(DM / BN, MR / BM);
  gemm_bf16_wmma<<<g1, 256, 0, stream>>>(x2, wqT, bq, nullptr, qb,   MR, DM, DM, GEMM_MODE_QKV);
  gemm_bf16_wmma<<<g1, 256, 0, stream>>>(x2, wkT, bk, nullptr, kbuf, MR, DM, DM, GEMM_MODE_QKV);
  gemm_bf16_wmma<<<g1, 256, 0, stream>>>(x2, wvT, bv, nullptr, vbuf, MR, DM, DM, GEMM_MODE_QKV_VT);
  dim3 ga(SS / QB, BB * NH);
  flash_attn_wmma<<<ga, 256, 0, stream>>>(qb, kbuf, vbuf, mask, attn);
  gemm_bf16_wmma<<<g1, 256, 0, stream>>>(attn, woT, bo, x, y1, MR, DM, DM, GEMM_MODE_RES);

  // ---- feedforward sublayer ----
  layernorm_bf16<<<MR, 256, 0, stream>>>(y1, a2, be2, x2);
  dim3 g2(DFF / BN, MR / BM);
  gemm_bf16_wmma<<<g2, 256, 0, stream>>>(x2, w1T, b1, nullptr, ffh, MR, DFF, DM, GEMM_MODE_RELU);
  dim3 g3(DM / BN, MR / BM);
  gemm_bf16_wmma<<<g3, 256, 0, stream>>>(ffh, w2T, b2, y1, (float*)d_out, MR, DM, DFF, GEMM_MODE_RES);
}
